// LlamaAttention_39479339385035
// MI455X (gfx1250) — matrix operand order repaired and checked by reading, not compiled, not run on hardware
//
#include <hip/hip_runtime.h>
#include <hip/hip_bf16.h>

// ---------------- Problem constants ----------------
constexpr int S_   = 2048;   // sequence length
constexpr int D_   = 4096;   // hidden dim
constexpr int H_   = 32;     // query heads
constexpr int KV_  = 8;      // kv heads
constexpr int HD_  = 128;    // head dim
constexpr int GRP_ = H_ / KV_;
constexpr float SCALING_ = 0.08838834764831845f;   // 1/sqrt(128)

// ---------------- CDNA5 WMMA types ----------------
typedef __attribute__((ext_vector_type(16))) __bf16 v16bf;
typedef __attribute__((ext_vector_type(8)))  float  v8f;

union Frag16 { v16bf v; uint4 u[2]; };

static __device__ __forceinline__ v8f zero8() {
  v8f z = {0.f,0.f,0.f,0.f,0.f,0.f,0.f,0.f};
  return z;
}

// float -> bf16, round-to-nearest-even
static __device__ __forceinline__ __bf16 f2bf(float f) {
  union { float f; unsigned int u; } x; x.f = f;
  unsigned int r = x.u + 0x7FFFu + ((x.u >> 16) & 1u);
  unsigned short h = (unsigned short)(r >> 16);
  return __builtin_bit_cast(__bf16, h);
}

static __device__ __forceinline__ v8f wmma_bf16(v16bf a, v16bf b, v8f c) {
  return __builtin_amdgcn_wmma_f32_16x16x32_bf16(
      /*neg_a=*/false, a, /*neg_b=*/false, b,
      /*c_mod=*/(short)0, c, /*reuse_a=*/false, /*reuse_b=*/false);
}

// A-fragment (16x32 bf16, MxK), src row-major with leading dim ld.
// Lane: M = row0 + l16.  half-wave 0: K 0-7 (v0-3), 16-23 (v4-7);
//                        half-wave 1: K 8-15, 24-31.   (ISA 7.12.2)
static __device__ __forceinline__ v16bf
load_a16x32(const __bf16* __restrict__ p, int ld, int row0, int k0, int l16, int half) {
  const __bf16* q = p + (size_t)(row0 + l16) * ld + k0 + half * 8;
  Frag16 f;
  f.u[0] = *(const uint4*)q;
  f.u[1] = *(const uint4*)(q + 16);
  return f.v;
}

// B-fragment (32x16 bf16, KxN). B[kk][n] = src[row0+n][k0+kk], src row-major ld.
// Lane: N = row0 + l16.  half-wave 0: K 0-7, 16-23; half-wave 1: K 8-15, 24-31
// (same K order as the A-fragment).
static __device__ __forceinline__ v16bf
load_b32x16(const __bf16* __restrict__ p, int ld, int row0, int k0, int l16, int half) {
  const __bf16* q = p + (size_t)(row0 + l16) * ld + k0 + half * 8;
  Frag16 f;
  f.u[0] = *(const uint4*)q;
  f.u[1] = *(const uint4*)(q + 16);
  return f.v;
}

static __device__ __forceinline__ float grp16_max(float x) {
  x = fmaxf(x, __shfl_xor(x, 1, 32));
  x = fmaxf(x, __shfl_xor(x, 2, 32));
  x = fmaxf(x, __shfl_xor(x, 4, 32));
  x = fmaxf(x, __shfl_xor(x, 8, 32));
  return x;
}
static __device__ __forceinline__ float grp16_sum(float x) {
  x += __shfl_xor(x, 1, 32);
  x += __shfl_xor(x, 2, 32);
  x += __shfl_xor(x, 4, 32);
  x += __shfl_xor(x, 8, 32);
  return x;
}

// ---------------- Prep kernels ----------------
__global__ void cvt_f32_to_bf16(const float* __restrict__ x, __bf16* __restrict__ y, int n) {
  int i = blockIdx.x * blockDim.x + threadIdx.x;
  if (i < n) y[i] = f2bf(x[i]);
}

// W[R][C] fp32 -> Wt[C][R] bf16
__global__ void transpose_to_bf16(const float* __restrict__ W, __bf16* __restrict__ Wt,
                                  int R, int C) {
  long long idx = (long long)blockIdx.x * blockDim.x + threadIdx.x;
  int k = (int)(idx % R);
  int n = (int)(idx / R);
  if (n < C) Wt[(size_t)n * R + k] = f2bf(W[(size_t)k * C + n]);
}

// RoPE: in fp32 [S][heads*128] -> out bf16 [heads][S][128]
__global__ void rope_to_bf16(const float* __restrict__ X,
                             const float* __restrict__ cosp,
                             const float* __restrict__ sinp,
                             __bf16* __restrict__ out, int heads) {
  int idx = blockIdx.x * blockDim.x + threadIdx.x;
  int total = S_ * heads * 64;
  if (idx >= total) return;
  int i = idx & 63;
  int h = (idx >> 6) % heads;
  int s = idx / (64 * heads);
  const float* row = X + (size_t)s * heads * HD_ + h * HD_;
  float x1 = row[i], x2 = row[i + 64];
  float c1 = cosp[s * HD_ + i],      s1 = sinp[s * HD_ + i];
  float c2 = cosp[s * HD_ + i + 64], s2 = sinp[s * HD_ + i + 64];
  __bf16* o = out + (size_t)h * S_ * HD_ + (size_t)s * HD_;
  o[i]      = f2bf(x1 * c1 - x2 * s1);
  o[i + 64] = f2bf(x2 * c2 + x1 * s2);
}

// V fp32 [S][KV*128] -> Vt bf16 [KV][128][S]
__global__ void v_transpose_bf16(const float* __restrict__ V, __bf16* __restrict__ Vt) {
  int idx = blockIdx.x * blockDim.x + threadIdx.x;
  int total = KV_ * HD_ * S_;
  if (idx >= total) return;
  int s  = idx % S_;
  int d  = (idx / S_) % HD_;
  int kv = idx / (S_ * HD_);
  Vt[(size_t)kv * HD_ * S_ + (size_t)d * S_ + s] =
      f2bf(V[(size_t)s * (KV_ * HD_) + kv * HD_ + d]);
}

// ---------------- WMMA GEMM: C[M][N] = A[M][K] * Bt[N][K]^T ----------------
// Each wave computes a 32(M) x 64(N) tile: 2 A-frags reuse 4 B-frags per
// k-step -> 8 WMMAs per 6 fragment loads (~21 FLOP/byte from L2).
// 8 waves / block.
__global__ __launch_bounds__(256) void gemm_bf16_wmma(
    const __bf16* __restrict__ A, const __bf16* __restrict__ Bt,
    float* __restrict__ C, int M, int N, int K) {
  int lane = threadIdx.x & 31;
  int wid  = threadIdx.x >> 5;
  int l16  = lane & 15;
  int half = lane >> 4;
  int tilesN = N >> 6;
  int gw = blockIdx.x * 8 + wid;
  int mt = gw / tilesN;
  int nt = gw % tilesN;
  int row0 = mt << 5;          // 32 rows per wave
  int col0 = nt << 6;          // 64 cols per wave

  v8f acc[8];
  #pragma unroll
  for (int i = 0; i < 8; ++i) acc[i] = zero8();

  for (int k0 = 0; k0 < K; k0 += 32) {
    __builtin_prefetch(A + (size_t)(row0 + l16) * K + k0 + 128, 0, 0);
    __builtin_prefetch(A + (size_t)(row0 + 16 + l16) * K + k0 + 128, 0, 0);
    v16bf a0 = load_a16x32(A, K, row0,      k0, l16, half);
    v16bf a1 = load_a16x32(A, K, row0 + 16, k0, l16, half);
    #pragma unroll
    for (int j = 0; j < 4; ++j) {
      v16bf b = load_b32x16(Bt, K, col0 + j * 16, k0, l16, half);
      acc[j]     = wmma_bf16(a0, b, acc[j]);
      acc[4 + j] = wmma_bf16(a1, b, acc[4 + j]);
    }
  }

  // C/D layout: lane holds column N=l16; VGPR v -> row M = v + half*8
  #pragma unroll
  for (int a = 0; a < 2; ++a) {
    #pragma unroll
    for (int v = 0; v < 8; ++v) {
      size_t r = (size_t)(row0 + a * 16 + v + half * 8) * N + col0 + l16;
      C[r]      = acc[a * 4 + 0][v];
      C[r + 16] = acc[a * 4 + 1][v];
      C[r + 32] = acc[a * 4 + 2][v];
      C[r + 48] = acc[a * 4 + 3][v];
    }
  }
}

// ---------------- Flash attention (causal, GQA) ----------------
// One wave per (head, 16-query tile); key tiles of 32; 4 waves per block.
__global__ __launch_bounds__(128) void flash_attn_wmma(
    const __bf16* __restrict__ Q,   // [H][S][128]  (RoPE'd)
    const __bf16* __restrict__ Kc,  // [KV][S][128] (RoPE'd)
    const __bf16* __restrict__ Vt,  // [KV][128][S]
    __bf16* __restrict__ Oattn) {   // [S][H*128]
  __shared__ __align__(16) __bf16 Plds[4][16][32];

  int lane = threadIdx.x & 31;
  int w    = threadIdx.x >> 5;
  int l16  = lane & 15;
  int half = lane >> 4;
  int qt   = blockIdx.x * 4 + w;
  int h    = blockIdx.y;
  int q0   = qt * 16;
  int kvh  = h / GRP_;

  const __bf16* Qh = Q  + (size_t)h   * S_ * HD_;
  const __bf16* Kh = Kc + (size_t)kvh * S_ * HD_;
  const __bf16* Vh = Vt + (size_t)kvh * HD_ * S_;

  v16bf qa[4];
  #pragma unroll
  for (int c = 0; c < 4; ++c)
    qa[c] = load_a16x32(Qh, HD_, q0, c * 32, l16, half);

  v8f o[8];
  #pragma unroll
  for (int d = 0; d < 8; ++d) o[d] = zero8();
  float m_i[8], l_i[8];
  #pragma unroll
  for (int v = 0; v < 8; ++v) { m_i[v] = -1e30f; l_i[v] = 0.f; }

  for (int t0 = 0; t0 < q0 + 16; t0 += 32) {
    // scores S = Q * K^T  (two 16x16 fragments: keys t0..t0+15, t0+16..t0+31)
    v8f s0 = zero8(), s1 = zero8();
    #pragma unroll
    for (int c = 0; c < 4; ++c) {
      v16bf b0 = load_b32x16(Kh, HD_, t0,      c * 32, l16, half);
      v16bf b1 = load_b32x16(Kh, HD_, t0 + 16, c * 32, l16, half);
      s0 = wmma_bf16(qa[c], b0, s0);
      s1 = wmma_bf16(qa[c], b1, s1);
    }

    int key0 = t0 + l16;
    int key1 = t0 + 16 + l16;
    #pragma unroll
    for (int v = 0; v < 8; ++v) {
      int qrow = q0 + v + half * 8;
      float a0 = s0[v] * SCALING_; if (key0 > qrow) a0 = -1e30f;
      float a1 = s1[v] * SCALING_; if (key1 > qrow) a1 = -1e30f;
      float rm   = grp16_max(fmaxf(a0, a1));
      float mnew = fmaxf(m_i[v], rm);
      float p0 = __expf(a0 - mnew);
      float p1 = __expf(a1 - mnew);
      float alpha = __expf(m_i[v] - mnew);
      float rs = grp16_sum(p0 + p1);
      l_i[v] = l_i[v] * alpha + rs;
      m_i[v] = mnew;
      #pragma unroll
      for (int d = 0; d < 8; ++d) o[d][v] *= alpha;
      // stage P (C-frag layout) into LDS for re-layout as A-frag
      Plds[w][v + half * 8][l16]      = f2bf(p0);
      Plds[w][v + half * 8][16 + l16] = f2bf(p1);
    }
    asm volatile("s_wait_dscnt 0" ::: "memory");

    // reload P as 16x32 A-fragment
    Frag16 pf;
    pf.u[0] = *(const uint4*)&Plds[w][l16][half * 8];
    pf.u[1] = *(const uint4*)&Plds[w][l16][16 + half * 8];
    v16bf pa = pf.v;

    // O += P * V   (8 dim-chunks of 16)
    #pragma unroll
    for (int d = 0; d < 8; ++d) {
      v16bf vb = load_b32x16(Vh, S_, d * 16, t0, l16, half);
      o[d] = wmma_bf16(pa, vb, o[d]);
    }
  }

  // normalize and emit bf16 attn output [S][H*128]
  #pragma unroll
  for (int d = 0; d < 8; ++d) {
    #pragma unroll
    for (int v = 0; v < 8; ++v) {
      float val = o[d][v] / l_i[v];
      Oattn[(size_t)(q0 + v + half * 8) * (H_ * HD_) + h * HD_ + d * 16 + l16] = f2bf(val);
    }
  }
}

// ---------------- Host launch ----------------
extern "C" void kernel_launch(void* const* d_in, const int* in_sizes, int n_in,
                              void* d_out, int out_size, void* d_ws, size_t ws_size,
                              hipStream_t stream) {
  const float* hs   = (const float*)d_in[0];  // (1,2048,4096)
  const float* cosp = (const float*)d_in[1];  // (1,2048,128)
  const float* sinp = (const float*)d_in[2];  // (1,2048,128)
  const float* Wq   = (const float*)d_in[3];  // (4096,4096)
  const float* Wk   = (const float*)d_in[4];  // (4096,1024)
  const float* Wv   = (const float*)d_in[5];  // (4096,1024)
  const float* Wo   = (const float*)d_in[6];  // (4096,4096)
  float* out = (float*)d_out;

  char* ws = (char*)d_ws;
  size_t off = 0;
  auto carve = [&](size_t bytes) { char* p = ws + off; off += (bytes + 255) & ~(size_t)255; return p; };

  __bf16* Xbf   = (__bf16*)carve((size_t)S_ * D_ * 2);
  __bf16* WqT   = (__bf16*)carve((size_t)D_ * (H_ * HD_) * 2);
  __bf16* WkT   = (__bf16*)carve((size_t)D_ * (KV_ * HD_) * 2);
  __bf16* WvT   = (__bf16*)carve((size_t)D_ * (KV_ * HD_) * 2);
  __bf16* WoT   = (__bf16*)carve((size_t)(H_ * HD_) * D_ * 2);
  float*  Qf32  = (float*)carve((size_t)S_ * (H_ * HD_) * 4);
  float*  Kf32  = (float*)carve((size_t)S_ * (KV_ * HD_) * 4);
  float*  Vf32  = (float*)carve((size_t)S_ * (KV_ * HD_) * 4);
  __bf16* Qbf   = (__bf16*)carve((size_t)H_ * S_ * HD_ * 2);
  __bf16* Kbf   = (__bf16*)carve((size_t)KV_ * S_ * HD_ * 2);
  __bf16* Vtbf  = (__bf16*)carve((size_t)KV_ * HD_ * S_ * 2);
  __bf16* Attn  = (__bf16*)carve((size_t)S_ * (H_ * HD_) * 2);
  (void)ws_size; (void)in_sizes; (void)n_in; (void)out_size;

  // 1) convert activations + transpose weights to bf16
  cvt_f32_to_bf16<<<(S_ * D_) / 256, 256, 0, stream>>>(hs, Xbf, S_ * D_);
  transpose_to_bf16<<<((size_t)D_ * 4096) / 256, 256, 0, stream>>>(Wq, WqT, D_, 4096);
  transpose_to_bf16<<<((size_t)D_ * 1024) / 256, 256, 0, stream>>>(Wk, WkT, D_, 1024);
  transpose_to_bf16<<<((size_t)D_ * 1024) / 256, 256, 0, stream>>>(Wv, WvT, D_, 1024);
  transpose_to_bf16<<<((size_t)4096 * D_) / 256, 256, 0, stream>>>(Wo, WoT, 4096, D_);

  // 2) QKV projections (WMMA GEMM, 32x64 wave tiles)
  gemm_bf16_wmma<<<(64 * 64) / 8, 256, 0, stream>>>(Xbf, WqT, Qf32, S_, 4096, D_);
  gemm_bf16_wmma<<<(64 * 16) / 8, 256, 0, stream>>>(Xbf, WkT, Kf32, S_, 1024, D_);
  gemm_bf16_wmma<<<(64 * 16) / 8, 256, 0, stream>>>(Xbf, WvT, Vf32, S_, 1024, D_);

  // 3) RoPE (Q, K) -> bf16 per-head layout; V -> transposed bf16
  rope_to_bf16<<<(S_ * H_ * 64) / 256, 256, 0, stream>>>(Qf32, cosp, sinp, Qbf, H_);
  rope_to_bf16<<<(S_ * KV_ * 64) / 256, 256, 0, stream>>>(Kf32, cosp, sinp, Kbf, KV_);
  v_transpose_bf16<<<(KV_ * HD_ * S_) / 256, 256, 0, stream>>>(Vf32, Vtbf);

  // 4) causal flash attention (WMMA)
  dim3 fgrid(S_ / 16 / 4, H_, 1);
  flash_attn_wmma<<<fgrid, 128, 0, stream>>>(Qbf, Kbf, Vtbf, Attn);

  // 5) output projection straight into d_out
  gemm_bf16_wmma<<<(64 * 64) / 8, 256, 0, stream>>>(Attn, WoT, out, S_, D_, H_ * HD_);
}
